// CompressionModule_14070312861857
// MI455X (gfx1250) — matrix operand order repaired and checked by reading, not compiled, not run on hardware
//
#include <hip/hip_runtime.h>

typedef __attribute__((ext_vector_type(16))) _Float16 v16h;
typedef __attribute__((ext_vector_type(8)))  _Float16 v8h;
typedef __attribute__((ext_vector_type(2)))  _Float16 h2;
typedef __attribute__((ext_vector_type(8)))  float    v8f;
typedef __attribute__((ext_vector_type(2)))  float    f2;

static constexpr int kRes[16] = {16,20,25,32,40,50,64,80,101,128,161,203,256,322,406,512};
static constexpr unsigned kPrime   = 2654435761u;
static constexpr unsigned kEncMask = 0x3FFFFu;   // N_ENC = 262144 = 2^18

struct Tabs { const float2* t[16]; };

// Same-wave LDS ordering: DS pipeline is in-order per wave; this pins the
// compiler's memory ordering and drains outstanding DS ops.
#define LDS_FENCE() asm volatile("s_wait_dscnt 0" ::: "memory")

__global__ void __launch_bounds__(128, 4)
ngp_mlp_fused(const float2* __restrict__ xn, Tabs tabs,
              const float* __restrict__ w0, const float* __restrict__ b0,
              const float* __restrict__ w1, const float* __restrict__ b1,
              const float* __restrict__ w2, const float* __restrict__ b2,
              const float* __restrict__ w3, const float* __restrict__ b3,
              float* __restrict__ out, int nGroups)
{
  const int lane = threadIdx.x & 31;
  const int wv   = threadIdx.x >> 5;       // wave within block (4 waves/block)
  const int n16  = lane & 15;
  const int half = lane >> 4;              // 0: lanes 0-15, 1: lanes 16-31
  const int waveId = (int)((blockIdx.x * blockDim.x + threadIdx.x) >> 5);
  const int nWaves = (int)((gridDim.x * blockDim.x) >> 5);

  // Per-wave double-buffers: two 16-point sub-tiles per iteration give two
  // independent WMMA chains to fill WMMA->VALU hazard windows.
  __shared__ alignas(16) _Float16 sFeat[4][2][16*32];
  __shared__ alignas(16) _Float16 sAct [4][2][16*64];
  // Block-shared pre-swizzled f16 B-fragments: 22 frags x 32 lanes x 16 f16.
  // Frag ids: 0-3 W0 | 4-7 W1.kLo | 8-11 W1.kHi | 12-15 W2.kLo | 16-19 W2.kHi
  //           20 W3.kLo | 21 W3.kHi
  __shared__ alignas(16) _Float16 sWB[22][32*16];

  // B k-order used here (contiguous, matches loadA): lane n = N, kb = (lane<16 ? 0 : 16),
  // vgpr j packs rows K = kb+2j, kb+2j+1.
  auto buildB = [&](const float* W, int ld, int k0, int nvalidN, int n0) -> v16h {
    v16h f;
    const int n  = n0 + n16;
    const int kb = k0 + half*16;
#pragma unroll
    for (int j = 0; j < 8; ++j) {
      float lo = (n < nvalidN) ? W[(kb + 2*j    )*ld + n] : 0.f;
      float hi = (n < nvalidN) ? W[(kb + 2*j + 1)*ld + n] : 0.f;
      f[2*j]   = (_Float16)lo;
      f[2*j+1] = (_Float16)hi;
    }
    return f;
  };

  // Cooperative one-time build of the shared fragment pool (wave wv takes
  // frags wv, wv+4, ...), then a single uniform block barrier.
  for (int f = wv; f < 22; f += 4) {
    v16h frag;
    if      (f < 4)  frag = buildB(w0, 64, 0,  64, f*16);
    else if (f < 8)  frag = buildB(w1, 64, 0,  64, (f-4)*16);
    else if (f < 12) frag = buildB(w1, 64, 32, 64, (f-8)*16);
    else if (f < 16) frag = buildB(w2, 64, 0,  64, (f-12)*16);
    else if (f < 20) frag = buildB(w2, 64, 32, 64, (f-16)*16);
    else if (f == 20) frag = buildB(w3, 3, 0,  3, 0);   // 64x3 zero-padded
    else              frag = buildB(w3, 3, 32, 3, 0);
    union { v16h v; v8h h[2]; } u; u.v = frag;
    *(v8h*)&sWB[f][lane*16]     = u.h[0];
    *(v8h*)&sWB[f][lane*16 + 8] = u.h[1];
  }
  __syncthreads();

  // Per-use fragment fetch: two ds_load_b128 from the shared pool.
  auto getB = [&](int f) -> v16h {
    union { v16h v; v8h h[2]; } u;
    u.h[0] = *(const v8h*)&sWB[f][lane*16];
    u.h[1] = *(const v8h*)&sWB[f][lane*16 + 8];
    return u.v;
  };

  // Packed-f16 per-column biases (1 VGPR each; bias is per-N = per-lane).
  h2 pb0[4], pb1[4], pb2[4];
#pragma unroll
  for (int t = 0; t < 4; ++t) {
    const _Float16 v0 = (_Float16)b0[t*16 + n16];
    const _Float16 v1 = (_Float16)b1[t*16 + n16];
    const _Float16 v2 = (_Float16)b2[t*16 + n16];
    pb0[t][0] = v0; pb0[t][1] = v0;
    pb1[t][0] = v1; pb1[t][1] = v1;
    pb2[t][0] = v2; pb2[t][1] = v2;
  }
  const float bi3 = (n16 < 3) ? b3[n16] : 0.f;

  // A k-order used here (contiguous, matches buildB): lane m = lane%16, kb = (lane<16 ? 0 : 16);
  // vgprs 0-3 = K kb..kb+7, vgprs 4-7 = K kb+8..kb+15  -> two b128 LDS loads.
  auto loadA = [&](const _Float16* base, int stride, int k0) -> v16h {
    union { v16h v; v8h h[2]; } u;
    u.h[0] = *(const v8h*)(base + n16*stride + k0 + half*16);
    u.h[1] = *(const v8h*)(base + n16*stride + k0 + half*16 + 8);
    return u.v;
  };

  // Retire one accumulator: packed f32->f16 cvt, +bias (pk_add), ReLU (pk_max),
  // then two b16 stores per pair.
  const h2 hz = { (_Float16)0, (_Float16)0 };
  auto storeAct = [&](int s, int t, const v8f& c, h2 bpk) {
#pragma unroll
    for (int r = 0; r < 8; r += 2) {
      f2 x = { c[r], c[r+1] };
      h2 y = __builtin_convertvector(x, h2);   // v_cvt_pk_f16_f32
      y = y + bpk;                             // v_pk_add_f16
      y = __builtin_elementwise_max(y, hz);    // v_pk_max_num_f16
      sAct[wv][s][(r     + half*8)*64 + t*16 + n16] = y[0];
      sAct[wv][s][(r + 1 + half*8)*64 + t*16 + n16] = y[1];
    }
  };

  const int nPairs = (nGroups + 1) >> 1;
  for (int p = waveId; p < nPairs; p += nWaves) {
    int grp[2];
    grp[0] = 2*p;
    grp[1] = (2*p + 1 < nGroups) ? (2*p + 1) : (nGroups - 1);

    // ---------- hash-grid encoding: 2 x (16 pts x 16 levels x 2 feats) ----------
#pragma unroll
    for (int s = 0; s < 2; ++s) {
      const float2 xy  = xn[grp[s]*16 + n16];
      const float  x01 = (xy.x + 1.f) * 0.5f;
      const float  y01 = (xy.y + 1.f) * 0.5f;
#pragma unroll
      for (int t = 0; t < 8; ++t) {
        const int     res = half ? kRes[2*t+1]   : kRes[2*t];
        const float2* tab = half ? tabs.t[2*t+1] : tabs.t[2*t];
        const float rm1 = (float)(res - 1);
        const float ux = x01 * rm1, uy = y01 * rm1;
        int ix0 = (int)floorf(ux);
        int iy0 = (int)floorf(uy);
        ix0 = ix0 < 0 ? 0 : (ix0 > res-2 ? res-2 : ix0);
        iy0 = iy0 < 0 ? 0 : (iy0 > res-2 ? res-2 : iy0);
        const float wx = ux - (float)ix0;
        const float wy = uy - (float)iy0;
        unsigned i00, i10, i01, i11;
        if (2*t + half >= 13) {                 // hashed levels 13..15
          const unsigned hy0 = (unsigned)iy0     * kPrime;
          const unsigned hy1 = (unsigned)(iy0+1) * kPrime;
          i00 = ((unsigned)ix0     ^ hy0) & kEncMask;
          i10 = ((unsigned)(ix0+1) ^ hy0) & kEncMask;
          i01 = ((unsigned)ix0     ^ hy1) & kEncMask;
          i11 = ((unsigned)(ix0+1) ^ hy1) & kEncMask;
        } else {                                // dense levels 0..12
          i00 = (unsigned)(iy0*res + ix0);
          i10 = i00 + 1u;
          i01 = i00 + (unsigned)res;
          i11 = i01 + 1u;
        }
        const float2 f00 = tab[i00], f10 = tab[i10], f01 = tab[i01], f11 = tab[i11];
        const float w00 = (1.f-wx)*(1.f-wy), w10 = wx*(1.f-wy);
        const float w01 = (1.f-wx)*wy,       w11 = wx*wy;
        const float e0 = f00.x*w00 + f10.x*w10 + f01.x*w01 + f11.x*w11;
        const float e1 = f00.y*w00 + f10.y*w10 + f01.y*w01 + f11.y*w11;
        union { _Float16 h[2]; unsigned u32; } pk;
        pk.h[0] = (_Float16)e0; pk.h[1] = (_Float16)e1;
        *(unsigned*)&sFeat[wv][s][n16*32 + 2*(2*t + half)] = pk.u32;
      }
    }
    LDS_FENCE();

    // ---------- layer 0: 2 x (16x32)@(32x64); C = inline 0, per-tile retire ----------
    {
      v16h a0[2];
      a0[0] = loadA(sFeat[wv][0], 32, 0);
      a0[1] = loadA(sFeat[wv][1], 32, 0);
#pragma unroll
      for (int t = 0; t < 4; ++t) {
        const v16h Bt = getB(t);
#pragma unroll
        for (int s = 0; s < 2; ++s) {
          v8f c = {};
          c = __builtin_amdgcn_wmma_f32_16x16x32_f16(false, a0[s], false, Bt,
                                                     (short)0, c, false, false);
          storeAct(s, t, c, pb0[t]);
        }
      }
    }
    LDS_FENCE();

    // ---------- layers 1 & 2: 2 x (16x64)@(64x64); per-tile retire ----------
#pragma unroll
    for (int L = 0; L < 2; ++L) {
      v16h aLo[2], aHi[2];
#pragma unroll
      for (int s = 0; s < 2; ++s) {
        aLo[s] = loadA(sAct[wv][s], 64, 0);
        aHi[s] = loadA(sAct[wv][s], 64, 32);
      }
      const int fLo = L ? 12 : 4, fHi = L ? 16 : 8;
#pragma unroll
      for (int t = 0; t < 4; ++t) {
        const v16h BtLo = getB(fLo + t);
        const v16h BtHi = getB(fHi + t);
        const h2 bpk = L ? pb2[t] : pb1[t];
#pragma unroll
        for (int s = 0; s < 2; ++s) {
          v8f c = {};
          c = __builtin_amdgcn_wmma_f32_16x16x32_f16(false, aLo[s], false, BtLo,
                                                     (short)0, c, false, false);
          c = __builtin_amdgcn_wmma_f32_16x16x32_f16(false, aHi[s], false, BtHi,
                                                     (short)0, c, false, false);
          storeAct(s, t, c, bpk);
        }
      }
      LDS_FENCE();
    }

    // ---------- layer 3: 2 x (16x64)@(64x16-padded), keep 3 cols ----------
    {
      v16h aLo[2], aHi[2];
#pragma unroll
      for (int s = 0; s < 2; ++s) {
        aLo[s] = loadA(sAct[wv][s], 64, 0);
        aHi[s] = loadA(sAct[wv][s], 64, 32);
      }
      const v16h B3a = getB(20);
      const v16h B3b = getB(21);
#pragma unroll
      for (int s = 0; s < 2; ++s) {
        v8f c = {};
        c = __builtin_amdgcn_wmma_f32_16x16x32_f16(false, aLo[s], false, B3a,
                                                   (short)0, c, false, false);
        c = __builtin_amdgcn_wmma_f32_16x16x32_f16(false, aHi[s], false, B3b,
                                                   (short)0, c, false, false);
        if (n16 < 3) {
#pragma unroll
          for (int r = 0; r < 8; ++r)
            out[(grp[s]*16 + r + half*8)*3 + n16] = c[r] + bi3;  // D: n=l%16, row r+8*(l/16)
        }
      }
    }
  }
}

extern "C" void kernel_launch(void* const* d_in, const int* in_sizes, int n_in,
                              void* d_out, int out_size, void* d_ws, size_t ws_size,
                              hipStream_t stream) {
  (void)n_in; (void)d_ws; (void)ws_size; (void)out_size;
  const float2* xn = (const float2*)d_in[0];
  Tabs tabs;
  for (int i = 0; i < 16; ++i) tabs.t[i] = (const float2*)d_in[1 + i];
  const float* w0 = (const float*)d_in[17];
  const float* b0 = (const float*)d_in[18];
  const float* w1 = (const float*)d_in[19];
  const float* b1 = (const float*)d_in[20];
  const float* w2 = (const float*)d_in[21];
  const float* b2 = (const float*)d_in[22];
  const float* w3 = (const float*)d_in[23];
  const float* b3 = (const float*)d_in[24];
  float* out = (float*)d_out;

  const int B = in_sizes[0] / 2;        // xn is (B,2)
  const int nGroups = B / 16;           // 16 points per WMMA M-tile
  dim3 grid(2048), block(128);          // 8192 waves; 32768 pairs -> 4 iters/wave
  hipLaunchKernelGGL(ngp_mlp_fused, grid, block, 0, stream,
                     xn, tabs, w0, b0, w1, b1, w2, b2, w3, b3, out, nGroups);
}
